// BiLSTMModule_41609643164217
// MI455X (gfx1250) — matrix operand order repaired and checked by reading, not compiled, not run on hardware
//
#include <hip/hip_runtime.h>
#include <hip/hip_bf16.h>

// BiLSTM: B=256, T=512, N=128, H=128. One persistent workgroup per
// (16-row batch tile, direction); weights resident in LDS as transposed bf16;
// per-step GEMMs via v_wmma_f32_16x16x32_bf16 with f32 accumulation.

#define BDIM 256
#define TDIM 512
#define NDIM 128
#define HDIM 128
#define GDIM 512          // 4*H
#define BM   16           // batch rows per workgroup
#define NTHREADS 512      // 16 waves of 32

typedef __attribute__((ext_vector_type(16))) __bf16 v16bf;
typedef __attribute__((ext_vector_type(8)))  float  v8f;

union Frag16 { uint4 q[2]; v16bf v; };

// A fragment (16x32 bf16, ISA 7.12.2): lane half h, row m = lane&15.
//   elems 0..7  = K k0+8h .. k0+8h+7      (VGPRs 0-3)
//   elems 8..15 = K k0+16+8h .. k0+16+8h+7 (VGPRs 4-7)
__device__ __forceinline__ v16bf load_a_frag(const __bf16* row, int k0, int half) {
    Frag16 f;
    f.q[0] = *(const uint4*)(row + k0 + 8 * half);
    f.q[1] = *(const uint4*)(row + k0 + 16 + 8 * half);
    return f.v;
}

// B fragment (32x16 bf16): lane n = lane&15, half h; elems 0..7 = K k0+8h+e, elems 8..15 = K k0+16+8h+(e-8).
// Weights stored transposed in LDS (col-major) so this is two contiguous 16-byte loads.
__device__ __forceinline__ v16bf load_b_frag(const __bf16* col, int k0, int half) {
    Frag16 f;
    f.q[0] = *(const uint4*)(col + k0 + 8 * half);
    f.q[1] = *(const uint4*)(col + k0 + 16 + 8 * half);
    return f.v;
}

__device__ __forceinline__ float fsig(float x)  { return 1.0f / (1.0f + __expf(-x)); }
__device__ __forceinline__ float ftanh(float x) { return 2.0f * fsig(2.0f * x) - 1.0f; }

__global__ __launch_bounds__(NTHREADS, 1)
void bilstm_kernel(const float* __restrict__ x,
                   const float* __restrict__ Wfw, const float* __restrict__ Ufw, const float* __restrict__ bfw,
                   const float* __restrict__ Wbw, const float* __restrict__ Ubw, const float* __restrict__ bbw,
                   float* __restrict__ out)
{
    extern __shared__ char smem[];
    __bf16* sWt = (__bf16*)smem;              // [GDIM][NDIM]  W transposed, bf16 (128 KB)
    __bf16* sUt = sWt + GDIM * NDIM;          // [GDIM][HDIM]  U transposed, bf16 (128 KB)
    __bf16* sX  = sUt + GDIM * NDIM;          // [BM][NDIM]    x_t tile (4 KB)
    __bf16* sH  = sX + BM * NDIM;             // [BM][HDIM]    h state bf16 (4 KB)
    float*  sC  = (float*)(sH + BM * HDIM);   // [BM][HDIM]    c state f32 (8 KB)
    float*  sZ  = sC + BM * HDIM;             // [BM][GDIM]    gate pre-acts (32 KB)
    float*  sB  = sZ + BM * GDIM;             // [GDIM]        bias (2 KB)   -> 313,344 B total

    const int tid  = threadIdx.x;
    const int lane = tid & 31;
    const int wave = tid >> 5;
    const int half = lane >> 4;
    const int nIdx = lane & 15;

    const int bTile = blockIdx.x;   // 0..15
    const int dir   = blockIdx.y;   // 0 = fwd, 1 = bwd

    const float* Wg = dir ? Wbw : Wfw;
    const float* Ug = dir ? Ubw : Ufw;
    const float* bg = dir ? bbw : bfw;

    // ---- one-time staging: transpose+quantize weights into LDS ----
    for (int i = tid; i < NDIM * GDIM; i += NTHREADS) {
        int n = i >> 9;            // row in W/U (coalesced global read)
        int c = i & (GDIM - 1);    // gate column
        sWt[c * NDIM + n] = (__bf16)Wg[i];
        sUt[c * NDIM + n] = (__bf16)Ug[i];
    }
    if (tid < GDIM) sB[tid] = bg[tid];
    for (int i = tid; i < BM * HDIM; i += NTHREADS) { sH[i] = (__bf16)0.0f; sC[i] = 0.0f; }
    __syncthreads();

    const int colBase = wave * 32;                  // 32 gate columns per wave
    const int mrow    = lane & 15;
    const __bf16* aRowX  = sX + mrow * NDIM;
    const __bf16* aRowH  = sH + mrow * HDIM;
    const __bf16* bCol0W = sWt + (colBase + nIdx) * NDIM;
    const __bf16* bCol1W = sWt + (colBase + 16 + nIdx) * NDIM;
    const __bf16* bCol0U = sUt + (colBase + nIdx) * NDIM;
    const __bf16* bCol1U = sUt + (colBase + 16 + nIdx) * NDIM;

    for (int t = 0; t < TDIM; ++t) {
        const int te = dir ? (TDIM - 1 - t) : t;    // backward cell reads reversed time

        // ---- stage x_t tile (f32 -> bf16), prefetch next step's rows ----
        for (int i = tid; i < BM * NDIM; i += NTHREADS) {
            int r = i >> 7;
            int n = i & (NDIM - 1);
            sX[i] = (__bf16)x[((size_t)(bTile * BM + r) * TDIM + te) * NDIM + n];
            if (t + 1 < TDIM) {
                int te2 = dir ? (TDIM - 2 - t) : (t + 1);
                __builtin_prefetch(x + ((size_t)(bTile * BM + r) * TDIM + te2) * NDIM + n, 0, 1);
            }
        }
        __syncthreads();

        // ---- z[16 x 32] = b + x_t@W + h@U  (this wave's column slice) ----
        float bias0 = sB[colBase + nIdx];
        float bias1 = sB[colBase + 16 + nIdx];
        v8f c0, c1;
        #pragma unroll
        for (int v = 0; v < 8; ++v) { c0[v] = bias0; c1[v] = bias1; }

        #pragma unroll
        for (int kc = 0; kc < 4; ++kc) {            // x_t @ W, K = 128
            int k0 = kc * 32;
            v16bf a = load_a_frag(aRowX, k0, half);
            c0 = __builtin_amdgcn_wmma_f32_16x16x32_bf16(false, a, false,
                     load_b_frag(bCol0W, k0, half), (short)0, c0, false, false);
            c1 = __builtin_amdgcn_wmma_f32_16x16x32_bf16(false, a, false,
                     load_b_frag(bCol1W, k0, half), (short)0, c1, false, false);
        }
        #pragma unroll
        for (int kc = 0; kc < 4; ++kc) {            // h @ U, K = 128
            int k0 = kc * 32;
            v16bf a = load_a_frag(aRowH, k0, half);
            c0 = __builtin_amdgcn_wmma_f32_16x16x32_bf16(false, a, false,
                     load_b_frag(bCol0U, k0, half), (short)0, c0, false, false);
            c1 = __builtin_amdgcn_wmma_f32_16x16x32_bf16(false, a, false,
                     load_b_frag(bCol1U, k0, half), (short)0, c1, false, false);
        }

        // ---- spill z tile to LDS (C/D layout: m = v + 8*half, n = lane&15) ----
        #pragma unroll
        for (int v = 0; v < 8; ++v) {
            int m = v + 8 * half;
            sZ[m * GDIM + colBase + nIdx]      = c0[v];
            sZ[m * GDIM + colBase + 16 + nIdx] = c1[v];
        }
        __syncthreads();

        // ---- LSTM cell update (gate order i, f, g, o) + output write ----
        for (int i = tid; i < BM * HDIM; i += NTHREADS) {
            int r = i >> 7;
            int j = i & (HDIM - 1);
            const float* zrow = sZ + r * GDIM;
            float ig = fsig(zrow[j]);
            float fg = fsig(zrow[HDIM + j]);
            float gg = ftanh(zrow[2 * HDIM + j]);
            float og = fsig(zrow[3 * HDIM + j]);
            float c  = fg * sC[i] + ig * gg;
            sC[i] = c;
            float h = og * ftanh(c);
            sH[i] = (__bf16)h;
            size_t b = (size_t)bTile * BM + r;
            // step-aligned concat: both streams written at loop index t
            out[(b * TDIM + t) * (2 * HDIM) + (size_t)dir * HDIM + j] = h;
        }
        __syncthreads();
    }
}

extern "C" void kernel_launch(void* const* d_in, const int* in_sizes, int n_in,
                              void* d_out, int out_size, void* d_ws, size_t ws_size,
                              hipStream_t stream) {
    const float* x   = (const float*)d_in[0];
    const float* Wf  = (const float*)d_in[1];
    const float* Uf  = (const float*)d_in[2];
    const float* bf_ = (const float*)d_in[3];
    const float* Wb  = (const float*)d_in[4];
    const float* Ub  = (const float*)d_in[5];
    const float* bb_ = (const float*)d_in[6];
    float* out = (float*)d_out;

    // LDS: 2*(512*128*2) + 2*(16*128*2) + 16*128*4 + 16*512*4 + 512*4 = 313,344 B (< 320 KB WGP LDS)
    size_t smem = (size_t)2 * GDIM * NDIM * 2 + (size_t)2 * BM * NDIM * 2
                + (size_t)BM * HDIM * 4 + (size_t)BM * GDIM * 4 + (size_t)GDIM * 4;

    dim3 grid(BDIM / BM, 2);   // 16 batch tiles x 2 directions = 32 persistent WGs
    bilstm_kernel<<<grid, dim3(NTHREADS), smem, stream>>>(x, Wf, Uf, bf_, Wb, Ub, bb_, out);
}
